// MultiHeadAttention_40054865002656
// MI455X (gfx1250) — matrix operand order repaired and checked by reading, not compiled, not run on hardware
//
#include <hip/hip_runtime.h>
#include <hip/hip_bf16.h>

// ---------------------------------------------------------------------------
// MHA forward for MI455X (gfx1250): bf16 WMMA everywhere, flash-attention
// tiling for the softmax(QK^T)V stage. wave32, 128-thread (4-wave) blocks.
// K/V tiles staged to LDS with GLOBAL_LOAD_ASYNC_TO_LDS_B128 (ASYNCcnt path)
// when the toolchain exposes the builtins; plain-copy fallback otherwise.
// ---------------------------------------------------------------------------

typedef __bf16 bf16_t;
typedef __attribute__((ext_vector_type(16))) __bf16 v16bf;
typedef __attribute__((ext_vector_type(8)))  float  v8f;

#define BATCH 4
#define SEQ   2048
#define DMODEL 768
#define NHEADS 12
#define HDIM  64
#define NTOK  (BATCH * SEQ)          // 8192

#if defined(__gfx1250__) &&                                                 \
    __has_builtin(__builtin_amdgcn_global_load_async_to_lds_b128) &&        \
    __has_builtin(__builtin_amdgcn_s_wait_asynccnt)
#define USE_ASYNC_LDS 1
#else
#define USE_ASYNC_LDS 0
#endif

#if USE_ASYNC_LDS
// Probe-derived prototype: param0 is 'int __attribute__((vector_size(16))) *'
// (generic pointer to 4xi32 payload); LDS side passed as addrspace(3) v4i*
// (implicitly convertible to generic if the param is generic).
typedef int v4i __attribute__((ext_vector_type(4)));
typedef __attribute__((address_space(3))) v4i v4i_lds;

static __device__ __forceinline__ void async_b128(void* lds, const void* g) {
  __builtin_amdgcn_global_load_async_to_lds_b128(
      (v4i*)g,
      (v4i_lds*)(unsigned)(unsigned long long)lds,   // low 32 bits = LDS offset
      /*imm offset=*/0, /*cpol=*/0);
}
#endif

static __device__ __forceinline__ v8f wmma_bf16(v16bf a, v16bf b, v8f c) {
  // D = A(16x32) * B(32x16) + C, f32 accumulate
  return __builtin_amdgcn_wmma_f32_16x16x32_bf16(
      /*neg_a=*/false, a, /*neg_b=*/false, b,
      /*c_mod=*/(short)0, c, /*reuse_a=*/false, /*reuse_b=*/false);
}

// ---- fragment builders (layouts per CDNA5 ISA 7.12.2, wave32) --------------
// A 16x32 bf16: lane L -> row M = L&15 ; elems 0..7 : K = base+j,
//                                  elems 8..15: K = 16+base+j, base = 8*(L>>4)
// B 32x16 bf16: lane L -> col N = L&15 ; same K order as A
static __device__ __forceinline__ v16bf frag_a_f32(const float* __restrict__ src,
                                                   int ld, int row0, int k0, int lane) {
  const int row  = row0 + (lane & 15);
  const int base = (lane >> 4) << 3;
  const float* p = src + (long)row * ld + k0 + base;
  v16bf r;
#pragma unroll
  for (int j = 0; j < 8; ++j) r[j] = (__bf16)p[j];
#pragma unroll
  for (int j = 0; j < 8; ++j) r[8 + j] = (__bf16)p[16 + j];
  return r;
}

static __device__ __forceinline__ v16bf frag_b_f32(const float* __restrict__ src,
                                                   int ld, int n0, int k0, int lane) {
  const int col  = n0 + (lane & 15);
  const float* p = src + (long)col * ld + k0 + ((lane >> 4) << 3);
  v16bf r;
#pragma unroll
  for (int j = 0; j < 16; ++j) r[j] = (__bf16)p[j < 8 ? j : j + 8];
  return r;
}

static __device__ __forceinline__ v16bf frag_a_bf16(const bf16_t* src,
                                                    int ld, int row0, int k0, int lane) {
  const int row  = row0 + (lane & 15);
  const int base = (lane >> 4) << 3;
  const bf16_t* p = src + (long)row * ld + k0 + base;
  v16bf r;
#pragma unroll
  for (int j = 0; j < 8; ++j) r[j] = p[j];
#pragma unroll
  for (int j = 0; j < 8; ++j) r[8 + j] = p[16 + j];
  return r;
}

static __device__ __forceinline__ v16bf frag_b_bf16(const bf16_t* src,
                                                    int ld, int n0, int k0, int lane) {
  const bf16_t* p = src + (long)(n0 + (lane & 15)) * ld + k0 + ((lane >> 4) << 3);
  v16bf r;
#pragma unroll
  for (int j = 0; j < 16; ++j) r[j] = p[j < 8 ? j : j + 8];
  return r;
}

// ---------------------------------------------------------------------------
// Kernel 1: fused QKV projection.  y = x @ W^T + b, written as bf16 in
// [b, h, s, hd] layout.  grid = (NTOK/64, DMODEL/64, 3); block = 128 (4 waves).
// ---------------------------------------------------------------------------
__global__ __launch_bounds__(128) void qkv_proj_kernel(
    const float* __restrict__ x,
    const float* __restrict__ Wq, const float* __restrict__ bq,
    const float* __restrict__ Wk, const float* __restrict__ bk,
    const float* __restrict__ Wv, const float* __restrict__ bv,
    bf16_t* __restrict__ Qws, bf16_t* __restrict__ Kws, bf16_t* __restrict__ Vws) {
  const int wave = threadIdx.x >> 5;
  const int lane = threadIdx.x & 31;
  const int m0   = blockIdx.x * 64 + wave * 16;
  const int n0   = blockIdx.y * 64;

  const float* W;
  const float* bias;
  bf16_t* dst;
  if (blockIdx.z == 0)      { W = Wq; bias = bq; dst = Qws; }
  else if (blockIdx.z == 1) { W = Wk; bias = bk; dst = Kws; }
  else                      { W = Wv; bias = bv; dst = Vws; }

  v8f acc0 = {}, acc1 = {}, acc2 = {}, acc3 = {};
  for (int k0 = 0; k0 < DMODEL; k0 += 32) {
    v16bf a = frag_a_f32(x, DMODEL, m0, k0, lane);
    v16bf b0 = frag_b_f32(W, DMODEL, n0 +  0, k0, lane);
    v16bf b1 = frag_b_f32(W, DMODEL, n0 + 16, k0, lane);
    v16bf b2 = frag_b_f32(W, DMODEL, n0 + 32, k0, lane);
    v16bf b3 = frag_b_f32(W, DMODEL, n0 + 48, k0, lane);
    acc0 = wmma_bf16(a, b0, acc0);
    acc1 = wmma_bf16(a, b1, acc1);
    acc2 = wmma_bf16(a, b2, acc2);
    acc3 = wmma_bf16(a, b3, acc3);
  }

  // D layout: lane L, vgpr r -> M = r + 8*(L>>4), N = L&15
  const int lh = lane >> 4, nc = lane & 15;
#pragma unroll
  for (int t = 0; t < 4; ++t) {
    const v8f& acc = (t == 0) ? acc0 : (t == 1) ? acc1 : (t == 2) ? acc2 : acc3;
    const int n = n0 + t * 16 + nc;
    const float bn = bias[n];
    const int h = n >> 6, hd = n & 63;
#pragma unroll
    for (int r = 0; r < 8; ++r) {
      const int m = m0 + r + 8 * lh;          // global token index
      const int b = m >> 11, s = m & 2047;
      dst[(((long)b * NHEADS + h) * SEQ + s) * HDIM + hd] = (__bf16)(acc[r] + bn);
    }
  }
}

// ---------------------------------------------------------------------------
// Kernel 2: flash attention.  grid = (SEQ/64, BATCH*NHEADS); block = 128.
// ---------------------------------------------------------------------------
__global__ __launch_bounds__(128) void attn_kernel(
    const bf16_t* __restrict__ Qws, const bf16_t* __restrict__ Kws,
    const bf16_t* __restrict__ Vws, bf16_t* __restrict__ Ows) {
  __shared__ __align__(16) bf16_t Kt[64 * 64];        // [key][hd]
  __shared__ __align__(16) bf16_t Vt[64 * 64];        // [hd][key]  (transposed)
#if USE_ASYNC_LDS
  __shared__ __align__(16) bf16_t Vs[64 * 64];        // [key][hd]  async stage
#endif
  __shared__ __align__(16) float  Sc[4][16 * 64];     // per-wave scores (f32)
  __shared__ __align__(16) bf16_t Pt[4][16 * 64];     // per-wave exp(P) (bf16)
  __shared__ float corr_s[4][16];
  __shared__ float linv_s[4][16];

  const int tid  = threadIdx.x;
  const int wave = tid >> 5;
  const int lane = tid & 31;
  const int bh   = blockIdx.y;                 // 0..47
  const int q0   = blockIdx.x * 64;
  const long base = (long)bh * SEQ * HDIM;
  const bf16_t* Qp = Qws + base;
  const bf16_t* Kp = Kws + base;
  const bf16_t* Vp = Vws + base;

  // Q fragments for this wave's 16 rows (held in VGPRs for the whole loop)
  const int mrow0 = q0 + wave * 16;
  const v16bf aq0 = frag_a_bf16(Qp, HDIM, mrow0, 0, lane);
  const v16bf aq1 = frag_a_bf16(Qp, HDIM, mrow0, 32, lane);

  v8f o0 = {}, o1 = {}, o2 = {}, o3 = {};
  float m_run = -3.0e38f, l_run = 0.0f;        // valid in lanes 0..15
  const int lh = lane >> 4, nc = lane & 15;

  for (int kb = 0; kb < SEQ; kb += 64) {
    const bf16_t* kgsrc = Kp + (long)kb * HDIM;   // contiguous 4096 elements
    const bf16_t* vgsrc = Vp + (long)kb * HDIM;

#if USE_ASYNC_LDS
    // ---- async stage: 8KB K tile + 8KB V tile, 4 b128 chunks per thread ----
#pragma unroll
    for (int i = 0; i < 4; ++i) {
      const int off = (tid + i * 128) * 8;        // elements (8 bf16 = 16B)
      async_b128(Kt + off, kgsrc + off);
      async_b128(Vs + off, vgsrc + off);
    }
    if (kb + 64 < SEQ) {                           // prefetch next tiles
      __builtin_prefetch(kgsrc + 64 * HDIM + tid * 32, 0, 3);
      __builtin_prefetch(vgsrc + 64 * HDIM + tid * 32, 0, 3);
    }
    __builtin_amdgcn_s_wait_asynccnt(0);
    __syncthreads();
    {   // transpose V stage (LDS -> LDS): Vt[hd][key] = Vs[key][hd]
      const int row  = tid >> 1;                  // key row 0..63
      const int half = (tid & 1) * 32;            // hd offset 0 or 32
      const bf16_t* vsrow = Vs + row * 64 + half;
#pragma unroll
      for (int i = 0; i < 32; ++i)
        Vt[(half + i) * 64 + row] = vsrow[i];
    }
    __syncthreads();
#else
    // ---- fallback: global -> VGPR -> LDS copy + transpose on the way in ----
    {
      const int row  = tid >> 1;                  // 0..63 key row
      const int half = (tid & 1) * 32;            // hd offset 0 or 32
      const bf16_t* kg = kgsrc + (long)row * HDIM + half;
      bf16_t* kd = Kt + row * 64 + half;
#pragma unroll
      for (int i = 0; i < 4; ++i)
        *(uint4*)(kd + i * 8) = *(const uint4*)(kg + i * 8);
      const bf16_t* vg = vgsrc + (long)row * HDIM + half;
#pragma unroll
      for (int i = 0; i < 32; ++i)
        Vt[(half + i) * 64 + row] = vg[i];
    }
    __syncthreads();
#endif

    // ---- scores S = Q * K^T (contraction over hd): 16x64 per wave ----
    v8f s0 = {}, s1 = {}, s2 = {}, s3 = {};
    {
      v16bf b;
      b = frag_b_bf16(Kt, 64,  0,  0, lane); s0 = wmma_bf16(aq0, b, s0);
      b = frag_b_bf16(Kt, 64,  0, 32, lane); s0 = wmma_bf16(aq1, b, s0);
      b = frag_b_bf16(Kt, 64, 16,  0, lane); s1 = wmma_bf16(aq0, b, s1);
      b = frag_b_bf16(Kt, 64, 16, 32, lane); s1 = wmma_bf16(aq1, b, s1);
      b = frag_b_bf16(Kt, 64, 32,  0, lane); s2 = wmma_bf16(aq0, b, s2);
      b = frag_b_bf16(Kt, 64, 32, 32, lane); s2 = wmma_bf16(aq1, b, s2);
      b = frag_b_bf16(Kt, 64, 48,  0, lane); s3 = wmma_bf16(aq0, b, s3);
      b = frag_b_bf16(Kt, 64, 48, 32, lane); s3 = wmma_bf16(aq1, b, s3);
    }

    // ---- spill scaled scores to LDS (1/sqrt(64) = 0.125) ----
    {
      float* sw = Sc[wave];
#pragma unroll
      for (int r = 0; r < 8; ++r) {
        const int row = (r + 8 * lh) * 64;
        sw[row + nc +  0] = s0[r] * 0.125f;
        sw[row + nc + 16] = s1[r] * 0.125f;
        sw[row + nc + 32] = s2[r] * 0.125f;
        sw[row + nc + 48] = s3[r] * 0.125f;
      }
    }
    __syncthreads();

    // ---- online softmax: lanes 0..15 each own one query row ----
    if (lane < 16) {
      const float* row = Sc[wave] + lane * 64;
      float bmax = row[0];
#pragma unroll
      for (int j = 1; j < 64; ++j) bmax = fmaxf(bmax, row[j]);
      const float mn = fmaxf(m_run, bmax);
      const float c  = __expf(m_run - mn);
      float sum = 0.0f;
      bf16_t* pd = Pt[wave] + lane * 64;
#pragma unroll
      for (int j = 0; j < 64; ++j) {
        const float e = __expf(row[j] - mn);
        sum += e;
        pd[j] = (__bf16)e;
      }
      l_run = l_run * c + sum;
      m_run = mn;
      corr_s[wave][lane] = c;
    }
    __syncthreads();

    // ---- rescale running O by per-row correction ----
#pragma unroll
    for (int r = 0; r < 8; ++r) {
      const float c = corr_s[wave][r + 8 * lh];
      o0[r] *= c; o1[r] *= c; o2[r] *= c; o3[r] *= c;
    }

    // ---- O += P * V (contraction over keys; B from transposed V tile) ----
    {
      const v16bf ap0 = frag_a_bf16(Pt[wave], 64, 0,  0, lane);
      const v16bf ap1 = frag_a_bf16(Pt[wave], 64, 0, 32, lane);
      v16bf b;
      b = frag_b_bf16(Vt, 64,  0,  0, lane); o0 = wmma_bf16(ap0, b, o0);
      b = frag_b_bf16(Vt, 64,  0, 32, lane); o0 = wmma_bf16(ap1, b, o0);
      b = frag_b_bf16(Vt, 64, 16,  0, lane); o1 = wmma_bf16(ap0, b, o1);
      b = frag_b_bf16(Vt, 64, 16, 32, lane); o1 = wmma_bf16(ap1, b, o1);
      b = frag_b_bf16(Vt, 64, 32,  0, lane); o2 = wmma_bf16(ap0, b, o2);
      b = frag_b_bf16(Vt, 64, 32, 32, lane); o2 = wmma_bf16(ap1, b, o2);
      b = frag_b_bf16(Vt, 64, 48,  0, lane); o3 = wmma_bf16(ap0, b, o3);
      b = frag_b_bf16(Vt, 64, 48, 32, lane); o3 = wmma_bf16(ap1, b, o3);
    }
    __syncthreads();   // before next iteration overwrites Kt/Vt
  }

  if (lane < 16) linv_s[wave][lane] = 1.0f / l_run;
  __syncthreads();

  // ---- normalize and write attn-out in [b, s, h*64+hd] bf16 layout ----
  const int b = bh / NHEADS, h = bh % NHEADS;
#pragma unroll
  for (int r = 0; r < 8; ++r) {
    const float inv = linv_s[wave][r + 8 * lh];
    const int s = q0 + wave * 16 + r + 8 * lh;
    const long rowo = ((long)b * SEQ + s) * DMODEL + h * HDIM;
    Ows[rowo + nc +  0] = (__bf16)(o0[r] * inv);
    Ows[rowo + nc + 16] = (__bf16)(o1[r] * inv);
    Ows[rowo + nc + 32] = (__bf16)(o2[r] * inv);
    Ows[rowo + nc + 48] = (__bf16)(o3[r] * inv);
  }
}

// ---------------------------------------------------------------------------
// Kernel 3: output projection.  out = attn @ OW^T + OW_b (f32 result).
// ---------------------------------------------------------------------------
__global__ __launch_bounds__(128) void oproj_kernel(
    const bf16_t* __restrict__ attn, const float* __restrict__ W,
    const float* __restrict__ bias, float* __restrict__ out) {
  const int wave = threadIdx.x >> 5;
  const int lane = threadIdx.x & 31;
  const int m0   = blockIdx.x * 64 + wave * 16;
  const int n0   = blockIdx.y * 64;

  v8f acc0 = {}, acc1 = {}, acc2 = {}, acc3 = {};
  for (int k0 = 0; k0 < DMODEL; k0 += 32) {
    v16bf a  = frag_a_bf16(attn, DMODEL, m0, k0, lane);
    v16bf b0 = frag_b_f32(W, DMODEL, n0 +  0, k0, lane);
    v16bf b1 = frag_b_f32(W, DMODEL, n0 + 16, k0, lane);
    v16bf b2 = frag_b_f32(W, DMODEL, n0 + 32, k0, lane);
    v16bf b3 = frag_b_f32(W, DMODEL, n0 + 48, k0, lane);
    acc0 = wmma_bf16(a, b0, acc0);
    acc1 = wmma_bf16(a, b1, acc1);
    acc2 = wmma_bf16(a, b2, acc2);
    acc3 = wmma_bf16(a, b3, acc3);
  }

  const int lh = lane >> 4, nc = lane & 15;
#pragma unroll
  for (int t = 0; t < 4; ++t) {
    const v8f& acc = (t == 0) ? acc0 : (t == 1) ? acc1 : (t == 2) ? acc2 : acc3;
    const int n = n0 + t * 16 + nc;
    const float bn = bias[n];
#pragma unroll
    for (int r = 0; r < 8; ++r) {
      const int m = m0 + r + 8 * lh;
      out[(long)m * DMODEL + n] = acc[r] + bn;
    }
  }
}

// ---------------------------------------------------------------------------
extern "C" void kernel_launch(void* const* d_in, const int* in_sizes, int n_in,
                              void* d_out, int out_size, void* d_ws, size_t ws_size,
                              hipStream_t stream) {
  const float* x    = (const float*)d_in[0];
  const float* QW_w = (const float*)d_in[1];
  const float* QW_b = (const float*)d_in[2];
  const float* KW_w = (const float*)d_in[3];
  const float* KW_b = (const float*)d_in[4];
  const float* VW_w = (const float*)d_in[5];
  const float* VW_b = (const float*)d_in[6];
  const float* OW_w = (const float*)d_in[7];
  const float* OW_b = (const float*)d_in[8];
  float* out = (float*)d_out;

  const size_t nElem = (size_t)BATCH * NHEADS * SEQ * HDIM;   // 6,291,456
  bf16_t* Qws = (bf16_t*)d_ws;
  bf16_t* Kws = Qws + nElem;
  bf16_t* Vws = Kws + nElem;
  bf16_t* Ows = Vws + nElem;   // [B, S, D] bf16

  dim3 g1(NTOK / 64, DMODEL / 64, 3);
  qkv_proj_kernel<<<g1, 128, 0, stream>>>(x, QW_w, QW_b, KW_w, KW_b, VW_w, VW_b,
                                          Qws, Kws, Vws);
  dim3 g2(SEQ / 64, BATCH * NHEADS);
  attn_kernel<<<g2, 128, 0, stream>>>(Qws, Kws, Vws, Ows);
  dim3 g3(NTOK / 64, DMODEL / 64);
  oproj_kernel<<<g3, 128, 0, stream>>>(Ows, OW_w, OW_b, out);
}
